// ConvAttnLSTMCell_17300128268687
// MI455X (gfx1250) — matrix operand order repaired and checked by reading, not compiled, not run on hardware
//
#include <hip/hip_runtime.h>
#include <hip/hip_bf16.h>

// ---------------------------------------------------------------------------
// ConvAttnLSTMCell for MI455X (gfx1250, wave32, WMMA).
// B=128, C=128, H=W=16, E=128, NH=8, MEM=8, K=3, HD=16, D=4096.
// Convs run as implicit GEMM with v_wmma_f32_16x16x32_bf16 (f32 accum),
// whole per-batch padded image staged in LDS (<=166KB of the 320KB WGP LDS).
// ---------------------------------------------------------------------------

typedef __attribute__((ext_vector_type(16))) __bf16 v16bf;
typedef __attribute__((ext_vector_type(8)))  float  v8f;

#define HW 256
#define BATCH 128
#define NHEAD 8
#define MEMS 8
#define DHEAD 4096

// ---------------------------------------------------------------------------
// Pack conv weights (Cout, Cin, 3, 3) f32 into per-lane WMMA A-fragments:
// dst[mtile][kstep][lane][16] bf16, kstep = kid*(Cin/32) + ciblk.
// Per-lane K mapping (contiguous, matches B reads):
//   lane = hi*16 + m,  element s holds K = hi*16 + s.
// ---------------------------------------------------------------------------
__global__ __launch_bounds__(256)
void pack_weights(const float* __restrict__ w, __bf16* __restrict__ dst,
                  int Cout, int Cin) {
    const int CB = Cin / 32;
    const int KS = 9 * CB;
    const int MT = Cout / 16;
    const long total = (long)MT * KS * 512;
    for (long g = (long)blockIdx.x * blockDim.x + threadIdx.x; g < total;
         g += (long)gridDim.x * blockDim.x) {
        const int idx16 = (int)(g & 15);
        const int lane  = (int)((g >> 4) & 31);
        const long rest = g >> 9;
        const int ks = (int)(rest % KS);
        const int mt = (int)(rest / KS);
        const int kid = ks / CB;
        const int cib = ks - kid * CB;
        const int hi = lane >> 4;
        const int j = idx16 >> 1, t = idx16 & 1;
        const int kk = hi * 16 + j * 2 + t;
        const int ci = cib * 32 + kk;
        const int co = mt * 16 + (lane & 15);
        dst[g] = (__bf16)w[((long)co * Cin + ci) * 9 + kid];
    }
}

// ---------------------------------------------------------------------------
// 3x3 SAME conv, one workgroup (8 waves) per batch image.
// LDS: padded image [18*18][CIN] bf16 (channel-innermost so B-fragments are
// contiguous 32B per lane).  A-fragments: prepacked, contiguous 32B per lane
// (stream from L2 — weights are reused across all 128 batches).
// Decomposition: out[co][p] = sum_{kid} W_kid(Cout x Cin) * ShiftedImg_kid.
// C/D accumulator layout (ISA): vgpr r, lane l -> (M = r + (l>>4)*8, N = l&15).
// ---------------------------------------------------------------------------
template<int CIN, int COUT, int C0>
__global__ __launch_bounds__(256)
void conv3x3_wmma(const float* __restrict__ src0, const float* __restrict__ src1,
                  const __bf16* __restrict__ apack, const float* __restrict__ bias,
                  const float* __restrict__ resid, float* __restrict__ out) {
    extern __shared__ __align__(32) __bf16 simg[];   // [324][CIN]
    const int b = blockIdx.x, tid = threadIdx.x;

    // zero the padded image, then fill interior (f32 -> bf16 convert)
    for (int i = tid; i < 324 * CIN; i += 256) simg[i] = (__bf16)0.0f;
    __syncthreads();
    for (int i = tid; i < 256 * CIN; i += 256) {
        const int ci = i >> 8, p = i & 255;
        float v;
        if (ci < C0) v = src0[((long)b * C0 + ci) * HW + p];
        else         v = src1[((long)b * (CIN - C0) + (ci - C0)) * HW + p];
        simg[(((p >> 4) + 1) * 18 + (p & 15) + 1) * CIN + ci] = (__bf16)v;
    }
    __syncthreads();

    const int wv = tid >> 5, lane = tid & 31;
    const int n = lane & 15, hi = lane >> 4;
    constexpr int MT = COUT / 16, CB = CIN / 32, KS = 9 * CB;

    for (int mt = wv; mt < MT; mt += 8) {
        for (int pg = 0; pg < 4; ++pg) {           // 4 groups of 4 row-tiles
            v8f acc0 = {}, acc1 = {}, acc2 = {}, acc3 = {};
            const __bf16* ap = apack + (long)mt * KS * 512 + lane * 16;
            for (int ks = 0; ks < KS; ++ks) {
                const int kid = ks / CB;
                const int cib = ks - kid * CB;
                const int dy = kid / 3, dx = kid - dy * 3;
                const v16bf a = *(const v16bf*)ap;  ap += 512;
                const int col = n + dx;
                const int kb  = cib * 32 + hi * 16;
                // ptile == output row y; pixel p = y*16 + n, x = n.
                const v16bf b0 = *(const v16bf*)(simg + ((pg*4+0+dy)*18 + col) * CIN + kb);
                const v16bf b1 = *(const v16bf*)(simg + ((pg*4+1+dy)*18 + col) * CIN + kb);
                const v16bf b2 = *(const v16bf*)(simg + ((pg*4+2+dy)*18 + col) * CIN + kb);
                const v16bf b3 = *(const v16bf*)(simg + ((pg*4+3+dy)*18 + col) * CIN + kb);
                acc0 = __builtin_amdgcn_wmma_f32_16x16x32_bf16(false, a, false, b0, (short)0, acc0, false, false);
                acc1 = __builtin_amdgcn_wmma_f32_16x16x32_bf16(false, a, false, b1, (short)0, acc1, false, false);
                acc2 = __builtin_amdgcn_wmma_f32_16x16x32_bf16(false, a, false, b2, (short)0, acc2, false, false);
                acc3 = __builtin_amdgcn_wmma_f32_16x16x32_bf16(false, a, false, b3, (short)0, acc3, false, false);
            }
            for (int r = 0; r < 8; ++r) {
                const int co = mt * 16 + hi * 8 + r;
                const float bs = bias[co];
                const long obase = ((long)b * COUT + co) * HW;
                const int p0 = (pg*4+0)*16 + n, p1 = (pg*4+1)*16 + n;
                const int p2 = (pg*4+2)*16 + n, p3 = (pg*4+3)*16 + n;
                float v0 = acc0[r] + bs, v1 = acc1[r] + bs;
                float v2 = acc2[r] + bs, v3 = acc3[r] + bs;
                if (resid) {
                    v0 += resid[obase + p0]; v1 += resid[obase + p1];
                    v2 += resid[obase + p2]; v3 += resid[obase + p3];
                }
                out[obase + p0] = v0; out[obase + p1] = v1;
                out[obase + p2] = v2; out[obase + p3] = v3;
            }
        }
    }
}

// ---------------------------------------------------------------------------
// LSTM gates: cc = (B, 5E, HW); gate g occupies channels [g*128, g*128+128).
// c1 = sigmoid(f)*c_cur + sigmoid(i)*tanh(g);  store o = sigmoid(o), a = sigmoid(a).
// ---------------------------------------------------------------------------
__global__ __launch_bounds__(256)
void gates_kernel(const float* __restrict__ cc, const float* __restrict__ c_cur,
                  float* __restrict__ c1, float* __restrict__ og, float* __restrict__ ag) {
    const long i = (long)blockIdx.x * 256 + threadIdx.x;   // over B*E*HW
    const long b = i >> 15, r = i & 32767;                 // r = e*256 + p
    const float* g5 = cc + b * (5L * 32768);
    const float xi = g5[r], xf = g5[32768 + r], xo = g5[65536 + r];
    const float xg = g5[98304 + r], xa = g5[131072 + r];
    const float ii = 1.f / (1.f + __expf(-xi));
    const float ff = 1.f / (1.f + __expf(-xf));
    c1[i] = ff * c_cur[i] + ii * tanhf(xg);
    og[i] = 1.f / (1.f + __expf(-xo));
    ag[i] = 1.f / (1.f + __expf(-xa));
}

// ---------------------------------------------------------------------------
// Fused rolling-memory slide + positional add + attention.
// One block per (b, nh).  Thread t owns pixel p=t of every head-dim row,
// i.e. d = i*256 + t for i in 0..15 (coalesced across the wave).
// Pass A streams concat_k/v ONCE: writes ret_k (+pos) / ret_v and
// accumulates q.k partials; LDS reduce -> softmax; pass B re-reads the
// just-written ret_v (L2-hot) for the weighted sum.
// ---------------------------------------------------------------------------
__global__ __launch_bounds__(256)
void attention_kernel(const float* __restrict__ kqv,
                      const float* __restrict__ ck, const float* __restrict__ cv,
                      const int* __restrict__ mask,
                      const float* __restrict__ pos_w, const float* __restrict__ pos_b,
                      float* __restrict__ rk, float* __restrict__ rv,
                      float* __restrict__ attn) {
    __shared__ float red[8 * 256];
    __shared__ float wts[8];
    const int bh = blockIdx.x, t = threadIdx.x;
    const int b = bh >> 3, nh = bh & 7;
    const float* kq = kqv + ((long)b * 384 + nh * 48) * HW;

    float ql[16], kn[16], vn[16];
    #pragma unroll
    for (int i = 0; i < 16; ++i) {
        kn[i] = kq[i * HW + t];
        ql[i] = kq[(16 + i) * HW + t] * 0.015625f;   // 1/sqrt(4096)
        vn[i] = kq[(32 + i) * HW + t];
    }

    const long kvbase = (long)bh * MEMS * DHEAD;
    for (int m = 0; m < MEMS; ++m) {
        float acc = 0.f;
        #pragma unroll
        for (int i = 0; i < 16; ++i) {
            const long d = (long)i * 256 + t;
            float kvv = (m < 7) ? ck[kvbase + (long)(m + 1) * DHEAD + d] : kn[i];
            kvv += pos_w[(long)m * 32768 + (long)nh * DHEAD + d];
            rk[kvbase + (long)m * DHEAD + d] = kvv;
            acc += ql[i] * kvv;
            const float vvv = (m < 7) ? cv[kvbase + (long)(m + 1) * DHEAD + d] : vn[i];
            rv[kvbase + (long)m * DHEAD + d] = vvv;
        }
        red[m * 256 + t] = acc;
    }
    __syncthreads();
    if (t < 8) {
        float s = 0.f;
        for (int j = 0; j < 256; ++j) s += red[t * 256 + j];
        // mask: -inf where masked; slot 7 is SET to +3.0 (overrides mask)
        const float mv = (t == 7) ? 3.0f : (mask[bh * 8 + t] ? -__builtin_inff() : 0.0f);
        wts[t] = s + mv + pos_b[t * 8 + nh];
    }
    __syncthreads();
    if (t == 0) {
        float mx = wts[0];
        for (int m = 1; m < 8; ++m) mx = fmaxf(mx, wts[m]);
        float e[8], sum = 0.f;
        for (int m = 0; m < 8; ++m) { e[m] = __expf(wts[m] - mx); sum += e[m]; }
        const float inv = 1.f / sum;
        for (int m = 0; m < 8; ++m) wts[m] = e[m] * inv;
    }
    __syncthreads();
    #pragma unroll
    for (int i = 0; i < 16; ++i) {
        const long d = (long)i * 256 + t;
        float ao = 0.f;
        for (int m = 0; m < MEMS; ++m) ao += wts[m] * rv[kvbase + (long)m * DHEAD + d];
        attn[((long)b * 128 + nh * 16 + i) * HW + t] = ao;   // (b, E, H, W)
    }
}

// ---------------------------------------------------------------------------
// LayerNorm over (E,H,W)=32768 per batch + final LSTM combine.
// ---------------------------------------------------------------------------
__global__ __launch_bounds__(256)
void final_kernel(const float* __restrict__ y, const float* __restrict__ c1,
                  const float* __restrict__ og, const float* __restrict__ ag,
                  const float* __restrict__ nw, const float* __restrict__ nb,
                  float* __restrict__ hout, float* __restrict__ cnout) {
    __shared__ float r1[256], r2[256];
    const int b = blockIdx.x, t = threadIdx.x;
    const float* yb = y + (long)b * 32768;
    float s = 0.f, s2 = 0.f;
    for (int i = 0; i < 128; ++i) {
        const float v = yb[i * 256 + t];
        s += v; s2 += v * v;
    }
    r1[t] = s; r2[t] = s2;
    __syncthreads();
    for (int st = 128; st > 0; st >>= 1) {
        if (t < st) { r1[t] += r1[t + st]; r2[t] += r2[t + st]; }
        __syncthreads();
    }
    const float mu  = r1[0] * (1.f / 32768.f);
    const float var = r2[0] * (1.f / 32768.f) - mu * mu;
    const float rs  = rsqrtf(var + 1e-5f);
    for (int i = 0; i < 128; ++i) {
        const long idx = (long)b * 32768 + i * 256 + t;
        const float o_ = (yb[i * 256 + t] - mu) * rs * nw[i * 256 + t] + nb[i * 256 + t];
        const float cn = c1[idx] + ag[idx] * tanhf(o_);
        hout[idx]  = og[idx] * tanhf(cn);
        cnout[idx] = cn;
    }
}

// ---------------------------------------------------------------------------
extern "C" void kernel_launch(void* const* d_in, const int* in_sizes, int n_in,
                              void* d_out, int out_size, void* d_ws, size_t ws_size,
                              hipStream_t stream) {
    (void)in_sizes; (void)n_in; (void)out_size; (void)ws_size;
    const float* input    = (const float*)d_in[0];
    const float* h_cur    = (const float*)d_in[1];
    const float* c_cur    = (const float*)d_in[2];
    const float* concat_k = (const float*)d_in[3];
    const float* concat_v = (const float*)d_in[4];
    const int*   attnmask = (const int*)  d_in[5];
    const float* main_w   = (const float*)d_in[6];
    const float* main_b   = (const float*)d_in[7];
    const float* proj_w   = (const float*)d_in[8];
    const float* proj_b   = (const float*)d_in[9];
    const float* out_w    = (const float*)d_in[10];
    const float* out_b    = (const float*)d_in[11];
    const float* norm_w   = (const float*)d_in[12];
    const float* norm_b   = (const float*)d_in[13];
    const float* pos_w    = (const float*)d_in[14];
    const float* pos_b    = (const float*)d_in[15];

    // output layout: h_next | c_next | ret_k | ret_v
    const long N_HE = (long)BATCH * 128 * HW;               // 4,194,304
    const long N_KV = (long)BATCH * NHEAD * MEMS * DHEAD;   // 33,554,432
    float* h_out = (float*)d_out;
    float* c_out = h_out + N_HE;
    float* rk    = h_out + 2 * N_HE;
    float* rv    = rk + N_KV;

    // workspace layout (floats)
    float* F     = (float*)d_ws;
    float* c1    = F;
    float* og    = F + N_HE;
    float* ag    = F + 2 * N_HE;
    float* attnb = F + 3 * N_HE;
    float* yb    = F + 4 * N_HE;
    float* cc    = F + 5 * N_HE;             // B*640*HW = 20,971,520 floats
    float* kqv   = cc;                        // aliases cc (disjoint lifetime)
    __bf16* ap1  = (__bf16*)(cc + 20971520L); // 40*72*512
    __bf16* ap2  = ap1 + 1474560L;            // 24*36*512
    __bf16* ap3  = ap2 + 442368L;             // 8*36*512

    // 1) pack conv weights into WMMA A-fragment layout (bf16)
    pack_weights<<<1024, 256, 0, stream>>>(main_w, ap1, 640, 256);
    pack_weights<<<512,  256, 0, stream>>>(proj_w, ap2, 384, 128);
    pack_weights<<<256,  256, 0, stream>>>(out_w,  ap3, 128, 128);

    // 2) main conv: concat(input, h_cur) -> cc (B, 640, HW)
    conv3x3_wmma<256, 640, 128><<<BATCH, 256, 324 * 256 * 2, stream>>>(
        input, h_cur, ap1, main_b, nullptr, cc);

    // 3) gates -> c1, o, a
    gates_kernel<<<(int)(N_HE / 256), 256, 0, stream>>>(cc, c_cur, c1, og, ag);

    // 4) proj conv: input -> kqv (B, 384, HW)   (reuses cc storage)
    conv3x3_wmma<128, 384, 128><<<BATCH, 256, 324 * 128 * 2, stream>>>(
        input, nullptr, ap2, proj_b, nullptr, kqv);

    // 5) fused K/V slide + attention -> ret_k, ret_v, attn buffer
    attention_kernel<<<BATCH * NHEAD, 256, 0, stream>>>(
        kqv, concat_k, concat_v, attnmask, pos_w, pos_b, rk, rv, attnb);

    // 6) out conv + residual(input) -> y
    conv3x3_wmma<128, 128, 128><<<BATCH, 256, 324 * 128 * 2, stream>>>(
        attnb, nullptr, ap3, out_b, input, yb);

    // 7) LayerNorm + final LSTM combine -> h_next, c_next
    final_kernel<<<BATCH, 256, 0, stream>>>(yb, c1, og, ag, norm_w, norm_b,
                                            h_out, c_out);
}
